// VOneBlock_86698209837353
// MI455X (gfx1250) — matrix operand order repaired and checked by reading, not compiled, not run on hardware
//
#include <hip/hip_runtime.h>
#include <stdint.h>

// ---------------------------------------------------------------------------
// Types for WMMA / async copy
// ---------------------------------------------------------------------------
typedef __attribute__((ext_vector_type(16))) __bf16 v16bf;
typedef __attribute__((ext_vector_type(8)))  float  v8f;
typedef __attribute__((ext_vector_type(4)))  int    v4i;
typedef __attribute__((address_space(1))) v4i gv4i;   // global int4
typedef __attribute__((address_space(3))) v4i lv4i;   // LDS int4

// ---------------------------------------------------------------------------
// Problem constants
// ---------------------------------------------------------------------------
#define BATCH   16
#define CIN     3
#define HIN     224
#define WIN     224
#define COUT2   512          // both convs fused (256 + 256)
#define HOUT    56
#define WOUT    56
#define SPAT    (HOUT*WOUT)          // 3136
#define NTOT    (BATCH*SPAT)         // 50176
#define KREAL   (CIN*25*25)          // 1875
#define KPAD    1888                 // 59 * 32
#define INV_SQRT2 0.70710678118654752f

// GEMM block tile
#define BM      128
#define BN      64
#define KT      32

// gfx1250 async global->LDS path (guarded so compile can't break)
#if __has_builtin(__builtin_amdgcn_global_load_async_to_lds_b128)
#define USE_ASYNC 1
#else
#define USE_ASYNC 0
#endif

static __device__ __forceinline__ void wait_async_zero() {
#if __has_builtin(__builtin_amdgcn_s_wait_asynccnt)
    __builtin_amdgcn_s_wait_asynccnt(0);
#else
    asm volatile("s_wait_asynccnt 0x0" ::: "memory");
#endif
}

static __device__ __forceinline__ unsigned short f32_to_bf16(float f) {
    union { float f; uint32_t u; } a; a.f = f;
    uint32_t u = a.u;
    u += 0x7FFFu + ((u >> 16) & 1u);   // round to nearest even
    return (unsigned short)(u >> 16);
}

// ---------------------------------------------------------------------------
// Kernel 0: pack weights -> bf16 A matrix [512][KPAD], K = c*625 + ky*25 + kx
// ---------------------------------------------------------------------------
__global__ __launch_bounds__(256) void pack_w_kernel(
    const float* __restrict__ w0, const float* __restrict__ w1,
    unsigned short* __restrict__ A)
{
    size_t i = (size_t)blockIdx.x * 256 + threadIdx.x;
    const size_t total = (size_t)COUT2 * KPAD;
    if (i >= total) return;
    int m = (int)(i / KPAD);
    int k = (int)(i - (size_t)m * KPAD);
    unsigned short v = 0;
    if (k < KREAL) {
        float f = (m < 256) ? w0[m * KREAL + k] : w1[(m - 256) * KREAL + k];
        v = f32_to_bf16(f);
    }
    A[i] = v;
}

// ---------------------------------------------------------------------------
// Kernel 1: implicit-GEMM conv via WMMA bf16.
//   S[m][n] = sum_k A[m][k] * im2col(x)[k][n]
//   n = b*3136 + oy*56 + ox ; k = c*625 + ky*25 + kx
//   Block tile 128(M) x 64(N); 8 waves as 4(M) x 2(N); each wave holds
//   2 A-frags x 2 B-frags -> 4 WMMAs/K-step with fragment reuse.
//   A tile staged with GLOBAL_LOAD_ASYNC_TO_LDS_B128 (ASYNCcnt), overlapped
//   with the VALU im2col gather for the B tile.
// ---------------------------------------------------------------------------
__global__ __launch_bounds__(256) void conv_wmma_kernel(
    const float* __restrict__ x,
    const unsigned short* __restrict__ A,
    float* __restrict__ S)
{
    __shared__ __align__(16) unsigned short Asm[BM][KT];   // [m][k]  8 KB
    __shared__ __align__(16) unsigned short Bsm[BN][KT];   // [n][k]  4 KB (transposed)

    const int tid  = threadIdx.x;
    const int lane = tid & 31;
    const int wave = tid >> 5;
    const int wm   = wave >> 1;        // 0..3 -> M offset wm*32
    const int wn   = wave & 1;         // 0..1 -> N offset wn*32
    const int m0   = blockIdx.y * BM;  // 4 blocks in y
    const int n0   = blockIdx.x * BN;  // 784 blocks in x

    // B-gather mapping: thread handles k-row kkB and 8 consecutive n's
    const int kkB = tid >> 3;          // 0..31
    const int nnB = (tid & 7) * 8;     // 0,8,...,56
    int bArr[8], oyArr[8], oxArr[8];
    #pragma unroll
    for (int j = 0; j < 8; ++j) {
        int n  = n0 + nnB + j;
        int b  = n / SPAT;
        int sp = n - b * SPAT;
        int oy = sp / WOUT;
        int ox = sp - oy * WOUT;
        bArr[j] = b; oyArr[j] = oy; oxArr[j] = ox;
    }

    v8f acc[4] = {};   // [mi*2 + ni]

    for (int k0 = 0; k0 < KPAD; k0 += KT) {
        // ---- stage A tile: 128x32 bf16 = 512 x 16B chunks, 2 per thread ----
        #pragma unroll
        for (int q = tid; q < (BM * KT / 8); q += 256) {
            int row = q >> 2;
            int col = (q & 3) * 8;
            const unsigned short* src = &A[(size_t)(m0 + row) * KPAD + k0 + col];
#if USE_ASYNC
            __builtin_amdgcn_global_load_async_to_lds_b128(
                (gv4i*)(void*)src,
                (lv4i*)(void*)&Asm[row][col],
                0, 0);
#else
            *(uint4*)&Asm[row][col] = *(const uint4*)src;
#endif
        }

        // ---- stage B tile (im2col gather, bf16, stored [n][k]) ----
        // overlaps with the in-flight async A copy
        {
            int k = k0 + kkB;
            unsigned short vals[8];
            if (k < KREAL) {
                int c   = k / 625;
                int rem = k - c * 625;
                int ky  = rem / 25;
                int kx  = rem - ky * 25;
                #pragma unroll
                for (int j = 0; j < 8; ++j) {
                    int iy = oyArr[j] * 4 - 12 + ky;
                    int ix = oxArr[j] * 4 - 12 + kx;
                    float v = 0.0f;
                    if (iy >= 0 && iy < HIN && ix >= 0 && ix < WIN)
                        v = x[(((size_t)bArr[j] * CIN + c) * HIN + iy) * WIN + ix];
                    vals[j] = f32_to_bf16(v);
                }
            } else {
                #pragma unroll
                for (int j = 0; j < 8; ++j) vals[j] = 0;
            }
            #pragma unroll
            for (int j = 0; j < 8; ++j) Bsm[nnB + j][kkB] = vals[j];
        }

#if USE_ASYNC
        wait_async_zero();
#endif
        __syncthreads();

        // ---- fragments (ISA 16-bit 16x32 layouts) ----
        // A: lanes 0-15 k=0..7 & 16..23 ; lanes 16-31 k=8..15 & 24..31
        union { uint4 q[2]; v16bf v; } fa[2], fbv[2];
        {
            int kbA = (lane < 16) ? 0 : 8;
            int kbB = (lane < 16) ? 0 : 8;
            #pragma unroll
            for (int i = 0; i < 2; ++i) {
                int mrow = wm * 32 + i * 16 + (lane & 15);
                fa[i].q[0] = *(const uint4*)&Asm[mrow][kbA];
                fa[i].q[1] = *(const uint4*)&Asm[mrow][kbA + 16];
                int ncol = wn * 32 + i * 16 + (lane & 15);
                fbv[i].q[0] = *(const uint4*)&Bsm[ncol][kbB];
                fbv[i].q[1] = *(const uint4*)&Bsm[ncol][kbB + 16];
            }
        }

        #pragma unroll
        for (int mi = 0; mi < 2; ++mi)
            #pragma unroll
            for (int ni = 0; ni < 2; ++ni)
                acc[mi * 2 + ni] = __builtin_amdgcn_wmma_f32_16x16x32_bf16(
                    false, fa[mi].v, false, fbv[ni].v,
                    (short)0, acc[mi * 2 + ni], false, false);

        __syncthreads();
    }

    // ---- write C (VGPR r: lanes<16 -> M=r, lanes>=16 -> M=r+8) ----
    #pragma unroll
    for (int mi = 0; mi < 2; ++mi) {
        #pragma unroll
        for (int ni = 0; ni < 2; ++ni) {
            int mo = m0 + wm * 32 + mi * 16 + ((lane >= 16) ? 8 : 0);
            int no = n0 + wn * 32 + ni * 16 + (lane & 15);
            const v8f& a = acc[mi * 2 + ni];
            #pragma unroll
            for (int r = 0; r < 8; ++r)
                S[(size_t)(mo + r) * NTOT + no] = a[r];
        }
    }
}

// ---------------------------------------------------------------------------
// Kernel 2: combine simple/complex cells -> h (NCHW [16][256][3136])
// ---------------------------------------------------------------------------
__global__ __launch_bounds__(256) void combine_kernel(
    const float* __restrict__ S, float* __restrict__ h)
{
    int m  = blockIdx.y;
    int b  = blockIdx.z;
    int sp = blockIdx.x * 256 + threadIdx.x;
    if (sp >= SPAT) return;
    size_t n = (size_t)b * SPAT + sp;
    float s0 = S[(size_t)m * NTOT + n];
    float v;
    if (m < 128) {
        v = s0 > 0.0f ? s0 : 0.0f;
    } else {
        float s1 = S[(size_t)(m + 256) * NTOT + n];
        v = sqrtf(s0 * s0 + s1 * s1) * INV_SQRT2;
    }
    v *= 25.0f;                                    // k_exc; already >= 0 (ReLU no-op)
    h[((size_t)b * 256 + m) * SPAT + sp] = v;
}

// ---------------------------------------------------------------------------
// Kernel 3: recurrent elementwise update  h = 10*net/(0.2+net), net = h+2w
// ---------------------------------------------------------------------------
__global__ __launch_bounds__(256) void update_kernel(
    float* __restrict__ h, const float* __restrict__ fb, int use_fb)
{
    int b = blockIdx.y;
    int i = blockIdx.x * 256 + threadIdx.x;        // 0..802815 exactly
    size_t idx = (size_t)b * (256 * SPAT) + i;
    float hv = h[idx];
    float w  = use_fb ? fb[i] : hv;
    float net = hv + 2.0f * w;
    h[idx] = 10.0f * net / (0.2f + net);
}

// ---------------------------------------------------------------------------
// Kernel 4: long-range 16x16 edge-padded conv stage
//   mode 0: d = relu(X[g*8+j] - X[g*8+j+2])
//   mode 1: d = relu(X[g*8+j] - nbuf[j-2][g])
//   nbuf[j][g][sp] = conv(d, filt[j])
// ---------------------------------------------------------------------------
__global__ __launch_bounds__(256) void lr_stage_kernel(
    const float* __restrict__ X,      // h batch 0: [256][3136]
    float* __restrict__ nbuf,         // [8][32][3136]
    const float* __restrict__ lrf,    // [8][16][16]
    int jbase, int mode)
{
    int g = blockIdx.x;               // 0..31
    int j = jbase + blockIdx.y;

    __shared__ float d[SPAT];
    __shared__ float f[256];

    const float* a = X + (size_t)(g * 8 + j) * SPAT;
    const float* s = (mode == 0)
        ? (X + (size_t)(g * 8 + j + 2) * SPAT)
        : (nbuf + ((size_t)(j - 2) * 32 + g) * SPAT);

    for (int i = threadIdx.x; i < SPAT; i += 256) {
        float v = a[i] - s[i];
        d[i] = v > 0.0f ? v : 0.0f;
    }
    f[threadIdx.x] = lrf[j * 256 + threadIdx.x];
    __syncthreads();

    for (int i = threadIdx.x; i < SPAT; i += 256) {
        int y  = i / WOUT;
        int xo = i - y * WOUT;
        float acc = 0.0f;
        #pragma unroll 4
        for (int dy = 0; dy < 16; ++dy) {
            int yy = y - 7 + dy; yy = yy < 0 ? 0 : (yy > 55 ? 55 : yy);
            const float* row = &d[yy * WOUT];
            const float* fr  = &f[dy * 16];
            #pragma unroll
            for (int dx = 0; dx < 16; ++dx) {
                int xx = xo - 7 + dx; xx = xx < 0 ? 0 : (xx > 55 ? 55 : xx);
                acc += row[xx] * fr[dx];
            }
        }
        nbuf[((size_t)j * 32 + g) * SPAT + i] = acc;
    }
}

// ---------------------------------------------------------------------------
// Kernel 5: channel Gaussian sigma=0.5 (5 taps, symmetric reflect in group of 8)
// ---------------------------------------------------------------------------
__global__ __launch_bounds__(256) void chan_gauss_kernel(
    const float* __restrict__ nbuf, float* __restrict__ netm)
{
    int c  = blockIdx.y;
    int g  = c >> 3, j = c & 7;
    int sp = blockIdx.x * 256 + threadIdx.x;
    if (sp >= SPAT) return;
    const float k1[5] = {2.63867e-4f, 0.106451f, 0.786571f, 0.106451f, 2.63867e-4f};
    float acc = 0.0f;
    #pragma unroll
    for (int t = 0; t < 5; ++t) {
        int jj = j + t - 2;
        if (jj < 0) jj = -jj - 1;
        if (jj > 7) jj = 15 - jj;
        acc += k1[t] * nbuf[((size_t)jj * 32 + g) * SPAT + sp];
    }
    netm[(size_t)c * SPAT + sp] = acc;
}

// ---------------------------------------------------------------------------
// Kernel 6: compute normalized sigma=8 Gaussian taps (65) once per launch
// ---------------------------------------------------------------------------
__global__ void init_gauss_kernel(float* __restrict__ k2)
{
    __shared__ float w[65];
    __shared__ float ssum;
    int t = threadIdx.x;
    if (t < 65) {
        float dd = (float)(t - 32) * 0.125f;       // /8
        w[t] = __expf(-0.5f * dd * dd);
    }
    __syncthreads();
    if (t == 0) { float s = 0.f; for (int i = 0; i < 65; ++i) s += w[i]; ssum = s; }
    __syncthreads();
    if (t < 65) k2[t] = w[t] / ssum;
}

// ---------------------------------------------------------------------------
// Kernels 7/8: separable sigma=8 Gaussian with symmetric reflection
// ---------------------------------------------------------------------------
__global__ __launch_bounds__(256) void gaussH_kernel(
    const float* __restrict__ in, float* __restrict__ out,
    const float* __restrict__ k2)
{
    int c  = blockIdx.y;
    int sp = blockIdx.x * 256 + threadIdx.x;
    if (sp >= SPAT) return;
    int y = sp / WOUT, xx = sp - y * WOUT;
    float acc = 0.0f;
    for (int t = 0; t < 65; ++t) {
        int yy = y + t - 32;
        if (yy < 0) yy = -yy - 1; else if (yy > 55) yy = 111 - yy;
        acc += k2[t] * in[(size_t)c * SPAT + yy * WOUT + xx];
    }
    out[(size_t)c * SPAT + sp] = acc;
}

__global__ __launch_bounds__(256) void gaussW_kernel(
    const float* __restrict__ in, float* __restrict__ out,
    const float* __restrict__ k2)
{
    int c  = blockIdx.y;
    int sp = blockIdx.x * 256 + threadIdx.x;
    if (sp >= SPAT) return;
    int y = sp / WOUT, x0 = sp - y * WOUT;
    float acc = 0.0f;
    for (int t = 0; t < 65; ++t) {
        int xx = x0 + t - 32;
        if (xx < 0) xx = -xx - 1; else if (xx > 55) xx = 111 - xx;
        acc += k2[t] * in[(size_t)c * SPAT + y * WOUT + xx];
    }
    out[(size_t)c * SPAT + sp] = acc;
}

// ---------------------------------------------------------------------------
// Kernel 9: feedback  fb = 0.001*x*(1+5*netp)/(0.2+2*netm)
// ---------------------------------------------------------------------------
__global__ __launch_bounds__(256) void fb_kernel(
    const float* __restrict__ X, const float* __restrict__ nbuf,
    const float* __restrict__ netm, float* __restrict__ fb)
{
    int c  = blockIdx.y;
    int g  = c >> 3, j = c & 7;
    int sp = blockIdx.x * 256 + threadIdx.x;
    if (sp >= SPAT) return;
    float netp = nbuf[((size_t)j * 32 + g) * SPAT + sp];
    float xv   = X[(size_t)c * SPAT + sp];
    fb[(size_t)c * SPAT + sp] =
        0.001f * xv * (1.0f + 5.0f * netp) / (0.2f + 2.0f * netm[(size_t)c * SPAT + sp]);
}

// ---------------------------------------------------------------------------
// Kernel 10: instance norm (population var) per (b,c)
// ---------------------------------------------------------------------------
__global__ __launch_bounds__(256) void instnorm_kernel(
    const float* __restrict__ h, float* __restrict__ out)
{
    int bc = blockIdx.x;                            // 0..4095
    const float* p = h + (size_t)bc * SPAT;
    float s = 0.0f, s2 = 0.0f;
    for (int i = threadIdx.x; i < SPAT; i += 256) {
        float v = p[i]; s += v; s2 += v * v;
    }
    __shared__ float rs[256], rq[256];
    rs[threadIdx.x] = s; rq[threadIdx.x] = s2;
    __syncthreads();
    for (int off = 128; off > 0; off >>= 1) {
        if (threadIdx.x < off) {
            rs[threadIdx.x] += rs[threadIdx.x + off];
            rq[threadIdx.x] += rq[threadIdx.x + off];
        }
        __syncthreads();
    }
    float mean = rs[0] * (1.0f / SPAT);
    float var  = rq[0] * (1.0f / SPAT) - mean * mean;
    float inv  = rsqrtf(var + 1e-5f);
    for (int i = threadIdx.x; i < SPAT; i += 256)
        out[(size_t)bc * SPAT + i] = (p[i] - mean) * inv;
}

// ---------------------------------------------------------------------------
// Workspace layout (bytes, 256-aligned)
// ---------------------------------------------------------------------------
#define OFF_A    ((size_t)0)                         // 512*1888*2   = 1,933,312
#define OFF_S    ((size_t)1933312)                   // 512*50176*4  = 102,760,448
#define OFF_H    ((size_t)104693760)                 // 16*256*3136*4= 51,380,224
#define OFF_FB   ((size_t)156073984)                 // 256*3136*4   = 3,211,264
#define OFF_NB   ((size_t)159285248)                 // 8*32*3136*4  = 3,211,264
#define OFF_NM   ((size_t)162496512)                 // 256*3136*4   = 3,211,264
#define OFF_TMP  ((size_t)165707776)                 // 256*3136*4   = 3,211,264
#define OFF_K2   ((size_t)168919040)                 // 65*4

extern "C" void kernel_launch(void* const* d_in, const int* in_sizes, int n_in,
                              void* d_out, int out_size, void* d_ws, size_t ws_size,
                              hipStream_t stream)
{
    const float* x   = (const float*)d_in[0];   // (16,3,224,224)
    const float* w0  = (const float*)d_in[1];   // (256,3,25,25)
    const float* w1  = (const float*)d_in[2];   // (256,3,25,25)
    const float* lrf = (const float*)d_in[3];   // (8,16,16)

    char* ws = (char*)d_ws;
    unsigned short* A  = (unsigned short*)(ws + OFF_A);
    float* S    = (float*)(ws + OFF_S);
    float* h    = (float*)(ws + OFF_H);
    float* fbuf = (float*)(ws + OFF_FB);
    float* nbuf = (float*)(ws + OFF_NB);
    float* netm = (float*)(ws + OFF_NM);
    float* tmp  = (float*)(ws + OFF_TMP);
    float* k2   = (float*)(ws + OFF_K2);
    float* out  = (float*)d_out;

    // weight pack + fused dual-conv implicit GEMM (WMMA)
    pack_w_kernel<<<(COUT2 * KPAD + 255) / 256, 256, 0, stream>>>(w0, w1, A);
    conv_wmma_kernel<<<dim3(NTOT / BN, COUT2 / BM), 256, 0, stream>>>(x, A, S);
    combine_kernel<<<dim3(13, 256, BATCH), 256, 0, stream>>>(S, h);
    init_gauss_kernel<<<1, 128, 0, stream>>>(k2);

    for (int it = 0; it < 4; ++it) {
        update_kernel<<<dim3(SPAT, BATCH), 256, 0, stream>>>(h, fbuf, it > 0 ? 1 : 0);
        if (it == 3) break;   // last iteration's feedback is dead — skip long-range
        lr_stage_kernel<<<dim3(32, 4), 256, 0, stream>>>(h, nbuf, lrf, 0, 0);
        lr_stage_kernel<<<dim3(32, 2), 256, 0, stream>>>(h, nbuf, lrf, 4, 1);
        lr_stage_kernel<<<dim3(32, 2), 256, 0, stream>>>(h, nbuf, lrf, 6, 1);
        chan_gauss_kernel<<<dim3(13, 256), 256, 0, stream>>>(nbuf, netm);
        gaussH_kernel<<<dim3(13, 256), 256, 0, stream>>>(netm, tmp, k2);
        gaussW_kernel<<<dim3(13, 256), 256, 0, stream>>>(tmp, netm, k2);
        fb_kernel<<<dim3(13, 256), 256, 0, stream>>>(h, nbuf, netm, fbuf);
    }

    instnorm_kernel<<<BATCH * 256, 256, 0, stream>>>(h, out);
}
